// ChessMultiHeadAttention_30442728194620
// MI455X (gfx1250) — matrix operand order repaired and checked by reading, not compiled, not run on hardware
//
#include <hip/hip_runtime.h>

// ---------------------------------------------------------------------------
// Types / helpers
// ---------------------------------------------------------------------------
typedef unsigned short u16;
typedef unsigned int   u32;
typedef unsigned long long u64;
typedef __attribute__((ext_vector_type(16))) __bf16 v16bf;
typedef __attribute__((ext_vector_type(8)))  float  v8f;

union Frag16 { v16bf v; u32 u[8]; u16 h[16]; };

__device__ __forceinline__ float bf2f(u16 b) {
  u32 u = ((u32)b) << 16;
  return __builtin_bit_cast(float, u);
}
__device__ __forceinline__ u16 f2bf(float f) {
  u32 u = __builtin_bit_cast(u32, f);
  u32 r = u + 0x7FFFu + ((u >> 16) & 1u);   // round-to-nearest-even
  return (u16)(r >> 16);
}

// CDNA5 16-bit A-fragment K index for u32 pair v (0..7), khalf = lane>>4.
__device__ __forceinline__ int a_kidx(int v, int khalf) {
  return ((v & 4) ? 16 : 0) + khalf * 8 + (v & 3) * 2;
}

// gfx1250 async copy: LDS[laddr] <= MEM[gaddr], 16 bytes, tracked by ASYNCcnt.
__device__ __forceinline__ void async_ld_b128(u32 laddr, const void* gaddr) {
  asm volatile("global_load_async_to_lds_b128 %0, %1, off"
               :: "v"(laddr), "v"((u64)gaddr) : "memory");
}
__device__ __forceinline__ void wait_async0() {
  asm volatile("s_wait_asynccnt 0x0" ::: "memory");
}
__device__ __forceinline__ u32 lds_addr(const void* p) {
  return (u32)(u64)p;   // low 32 bits of generic shared pointer = LDS offset
}

// ---------------------------------------------------------------------------
// f32 -> bf16 conversion
// ---------------------------------------------------------------------------
__global__ __launch_bounds__(256) void f32_to_bf16_kernel(
    const float* __restrict__ in, u16* __restrict__ out, long long n) {
  long long i = (long long)blockIdx.x * blockDim.x + threadIdx.x;
  long long stride = (long long)gridDim.x * blockDim.x;
  for (; i < n; i += stride) out[i] = f2bf(in[i]);
}

// ---------------------------------------------------------------------------
// Generic bf16 WMMA GEMM: C[M,N] = A[M,K] @ B[K,N] (+bias, opt ReLU)
// 256 threads = 8 waves (2x4). Block tile 128x128, K-step 32.
// Double-buffered LDS; A-tile staged via GLOBAL_LOAD_ASYNC_TO_LDS_B128,
// B-tile via global->reg->transposed-scatter deferred past the WMMA block.
// Requires: M % 128 == 0, K % 32 == 0, N % 8 == 0 (N<128 via guards).
// ---------------------------------------------------------------------------
#define GEMM_BM 128
#define GEMM_BN 128
#define GEMM_BK 32
#define LPAD    16   // row stride 48 u16 = 96B: 16B aligned, conflict-reduced

template <typename OutT, bool RELU>
__global__ __launch_bounds__(256) void gemm_bf16_kernel(
    const u16* __restrict__ A, const u16* __restrict__ Bm,
    const float* __restrict__ bias, OutT* __restrict__ C,
    int M, int N, int K,
    long long aBatch, long long bBatch, long long biasBatch, long long cBatch) {
  __shared__ u16 sA [2][GEMM_BM][GEMM_BK + LPAD];  // [buf][m][k]
  __shared__ u16 sBT[2][GEMM_BN][GEMM_BK + LPAD];  // [buf][n][k]

  const int z = blockIdx.z;
  A  += (long long)z * aBatch;
  Bm += (long long)z * bBatch;
  const float* biasz = bias ? bias + (long long)z * biasBatch : nullptr;
  C  += (long long)z * cBatch;

  const int tid   = threadIdx.x;
  const int lane  = tid & 31;
  const int wid   = tid >> 5;
  const int waveM = wid >> 2;   // 0..1
  const int waveN = wid & 3;    // 0..3
  const int r16   = lane & 15;
  const int khalf = lane >> 4;
  const int mBase = blockIdx.y * GEMM_BM;
  const int nBase = blockIdx.x * GEMM_BN;

  // per-thread fixed load coordinates (2 x 16B chunks each for A and B)
  const int aRow0 = (tid * 2) >> 2,     aCol0 = ((tid * 2) & 3) * 8;
  const int aRow1 = (tid * 2 + 1) >> 2, aCol1 = ((tid * 2 + 1) & 3) * 8;
  const int bRow0 = (tid * 2) >> 4,     bCol0 = ((tid * 2) & 15) * 8;
  const int bRow1 = (tid * 2 + 1) >> 4, bCol1 = ((tid * 2 + 1) & 15) * 8;

  auto issueA = [&](int k0, int buf) {
    async_ld_b128(lds_addr(&sA[buf][aRow0][aCol0]),
                  A + (size_t)(mBase + aRow0) * K + k0 + aCol0);
    async_ld_b128(lds_addr(&sA[buf][aRow1][aCol1]),
                  A + (size_t)(mBase + aRow1) * K + k0 + aCol1);
  };
  auto loadB = [&](int k0, uint4* r) {
    r[0] = make_uint4(0u, 0u, 0u, 0u);
    r[1] = make_uint4(0u, 0u, 0u, 0u);
    if (nBase + bCol0 < N)
      r[0] = *(const uint4*)(Bm + (size_t)(k0 + bRow0) * N + nBase + bCol0);
    if (nBase + bCol1 < N)
      r[1] = *(const uint4*)(Bm + (size_t)(k0 + bRow1) * N + nBase + bCol1);
  };
  auto scatterB = [&](const uint4* r, int buf) {
    const u16* p0 = (const u16*)&r[0];
    const u16* p1 = (const u16*)&r[1];
#pragma unroll
    for (int j = 0; j < 8; ++j) sBT[buf][bCol0 + j][bRow0] = p0[j];
#pragma unroll
    for (int j = 0; j < 8; ++j) sBT[buf][bCol1 + j][bRow1] = p1[j];
  };

  v8f acc[4][2] = {};
  const int nK = K / GEMM_BK;

  // prologue: stage tile 0
  uint4 breg[2];
  loadB(0, breg);
  issueA(0, 0);
  scatterB(breg, 0);
  wait_async0();
  __syncthreads();

  for (int t = 0; t < nK; ++t) {
    const int cur = t & 1;
    const int nxt = cur ^ 1;
    const bool hasNext = (t + 1) < nK;
    if (hasNext) {
      loadB((t + 1) * GEMM_BK, breg);   // global loads in flight during WMMAs
      issueA((t + 1) * GEMM_BK, nxt);   // async copies in flight during WMMAs
    }
    if (t + 2 < nK)
      __builtin_prefetch(
          (const void*)(Bm + (size_t)((t + 2) * GEMM_BK + bRow0) * N + nBase + bCol0),
          0, 1);

    Frag16 af[4], bfr[2];
#pragma unroll
    for (int mt = 0; mt < 4; ++mt) {
      int row = waveM * 64 + mt * 16 + r16;
#pragma unroll
      for (int v = 0; v < 8; ++v)
        af[mt].u[v] = *(const u32*)&sA[cur][row][a_kidx(v, khalf)];
    }
#pragma unroll
    for (int nt = 0; nt < 2; ++nt) {
      int n = waveN * 32 + nt * 16 + r16;
#pragma unroll
      for (int v = 0; v < 8; ++v)
        bfr[nt].u[v] = *(const u32*)&sBT[cur][n][a_kidx(v, khalf)];
    }
#pragma unroll
    for (int mt = 0; mt < 4; ++mt)
#pragma unroll
      for (int nt = 0; nt < 2; ++nt)
        acc[mt][nt] = __builtin_amdgcn_wmma_f32_16x16x32_bf16(
            false, af[mt].v, false, bfr[nt].v, (short)0, acc[mt][nt], false, false);

    if (hasNext) scatterB(breg, nxt);   // waits its global loads only here
    wait_async0();
    __syncthreads();
  }

  // epilogue: C layout VGPR r -> M = khalf*8 + r, N = lane&15
#pragma unroll
  for (int mt = 0; mt < 4; ++mt) {
#pragma unroll
    for (int nt = 0; nt < 2; ++nt) {
      int n = nBase + waveN * 32 + nt * 16 + r16;
      if (n < N) {
        float bv = biasz ? biasz[n] : 0.0f;
        int m0 = mBase + waveM * 64 + mt * 16 + khalf * 8;
#pragma unroll
        for (int r = 0; r < 8; ++r) {
          float val = acc[mt][nt][r] + bv;
          if constexpr (RELU) val = fmaxf(val, 0.0f);
          if constexpr (sizeof(OutT) == 2)
            *(u16*)(C + (size_t)(m0 + r) * N + n) = f2bf(val);
          else
            *(float*)(C + (size_t)(m0 + r) * N + n) = val;
        }
      }
    }
  }
}

// ---------------------------------------------------------------------------
// LayerNorm over COMP=256, one row per block, 256 threads, bf16 out
// ---------------------------------------------------------------------------
__global__ __launch_bounds__(256) void layernorm_kernel(
    const float* __restrict__ x, const float* __restrict__ g,
    const float* __restrict__ bt, u16* __restrict__ out) {
  __shared__ float red[8];
  const int row  = blockIdx.x;
  const int tid  = threadIdx.x;
  const int lane = tid & 31;
  const int wid  = tid >> 5;
  float v = x[(size_t)row * 256 + tid];
  float s = v;
#pragma unroll
  for (int o = 16; o > 0; o >>= 1) s += __shfl_down(s, o);
  if (lane == 0) red[wid] = s;
  __syncthreads();
  float mu = 0.0f;
#pragma unroll
  for (int i = 0; i < 8; ++i) mu += red[i];
  mu *= (1.0f / 256.0f);
  __syncthreads();
  float d  = v - mu;
  float s2 = d * d;
#pragma unroll
  for (int o = 16; o > 0; o >>= 1) s2 += __shfl_down(s2, o);
  if (lane == 0) red[wid] = s2;
  __syncthreads();
  float var = 0.0f;
#pragma unroll
  for (int i = 0; i < 8; ++i) var += red[i];
  var *= (1.0f / 256.0f);
  float y = d * rsqrtf(var + 1e-5f) * g[tid] + bt[tid];
  out[(size_t)row * 256 + tid] = f2bf(y);
}

// ---------------------------------------------------------------------------
// Fused attention: one wave32 per (b,h). S=64, D=32.
// Q/K tiles staged via async-to-LDS; V transposed via reg scatter.
// scores = scale*Q@K^T + pos_bias + smolgen ; softmax ; out = P@V
// ---------------------------------------------------------------------------
__global__ __launch_bounds__(32) void attention_kernel(
    const u16* __restrict__ qg, const u16* __restrict__ kg,
    const u16* __restrict__ vg, const float* __restrict__ pos_bias,
    const u16* __restrict__ logits, u16* __restrict__ outg) {
  __shared__ u16   sQ [64][48];   // [s][d]
  __shared__ u16   sK [64][48];   // [s][d]
  __shared__ u16   sVT[32][80];   // [d][s]  (v transposed)
  __shared__ float sS [64][66];   // scores f32
  __shared__ u16   sP [64][80];   // probs bf16

  const int bh    = blockIdx.x;
  const int b     = bh >> 5;
  const int h     = bh & 31;
  const int lane  = threadIdx.x;
  const int r16   = lane & 15;
  const int khalf = lane >> 4;

  const u16* qbase = qg + ((size_t)b * 64) * 1024 + h * 32;
  const u16* kbase = kg + ((size_t)b * 64) * 1024 + h * 32;
  const u16* vbase = vg + ((size_t)b * 64) * 1024 + h * 32;

#pragma unroll
  for (int i = 0; i < 8; ++i) {
    int u     = lane + 32 * i;
    int row   = u >> 2;
    int chunk = (u & 3) * 8;
    async_ld_b128(lds_addr(&sQ[row][chunk]), qbase + (size_t)row * 1024 + chunk);
    async_ld_b128(lds_addr(&sK[row][chunk]), kbase + (size_t)row * 1024 + chunk);
    uint4 vv = *(const uint4*)(vbase + (size_t)row * 1024 + chunk);
    const u16* pvv = (const u16*)&vv;
#pragma unroll
    for (int j = 0; j < 8; ++j) sVT[chunk + j][row] = pvv[j];
  }
  wait_async0();
  __syncthreads();

  // ---- scores = Q @ K^T (D=32 -> single WMMA K-step) ----
  Frag16 qf[4], kf[4];
#pragma unroll
  for (int t = 0; t < 4; ++t) {
    int row = t * 16 + r16;
#pragma unroll
    for (int v = 0; v < 8; ++v) qf[t].u[v] = *(const u32*)&sQ[row][a_kidx(v, khalf)];
#pragma unroll
    for (int v = 0; v < 8; ++v) kf[t].u[v] = *(const u32*)&sK[row][a_kidx(v, khalf)];
  }
  const float scale = 0.17677669529663687f;  // 1/sqrt(32)
  const float* pb = pos_bias + (size_t)h * 64 * 64;
  const u16*   lg = logits + ((size_t)h * 1024 + b) * 4096;  // [H,B,S,S]
#pragma unroll
  for (int mt = 0; mt < 4; ++mt) {
#pragma unroll
    for (int nt = 0; nt < 4; ++nt) {
      v8f c = {};
      c = __builtin_amdgcn_wmma_f32_16x16x32_bf16(
          false, qf[mt].v, false, kf[nt].v, (short)0, c, false, false);
      int n  = nt * 16 + r16;
      int m0 = mt * 16 + khalf * 8;
#pragma unroll
      for (int r = 0; r < 8; ++r) {
        int m = m0 + r;
        sS[m][n] = c[r] * scale + pb[m * 64 + n] + bf2f(lg[m * 64 + n]);
      }
    }
  }
  __syncthreads();

  // ---- softmax: 2 rows per lane ----
#pragma unroll
  for (int t = 0; t < 2; ++t) {
    int m = lane * 2 + t;
    float mx = -3.0e38f;
    for (int j = 0; j < 64; ++j) mx = fmaxf(mx, sS[m][j]);
    float sum = 0.0f;
    for (int j = 0; j < 64; ++j) sum += __expf(sS[m][j] - mx);
    float inv = 1.0f / sum;
    for (int j = 0; j < 64; ++j) sP[m][j] = f2bf(__expf(sS[m][j] - mx) * inv);
  }
  __syncthreads();

  // ---- out = P @ V  (K=64 -> 2 WMMA K-steps; N=32 -> 2 tiles) ----
#pragma unroll
  for (int mt = 0; mt < 4; ++mt) {
#pragma unroll
    for (int nt = 0; nt < 2; ++nt) {
      v8f c = {};
#pragma unroll
      for (int kt = 0; kt < 2; ++kt) {
        Frag16 pf, vf;
#pragma unroll
        for (int v = 0; v < 8; ++v)
          pf.u[v] = *(const u32*)&sP[mt * 16 + r16][kt * 32 + a_kidx(v, khalf)];
#pragma unroll
        for (int v = 0; v < 8; ++v)
          vf.u[v] = *(const u32*)&sVT[nt * 16 + r16][kt * 32 + a_kidx(v, khalf)];
        c = __builtin_amdgcn_wmma_f32_16x16x32_bf16(
            false, pf.v, false, vf.v, (short)0, c, false, false);
      }
      int n  = nt * 16 + r16;
      int m0 = mt * 16 + khalf * 8;
#pragma unroll
      for (int r = 0; r < 8; ++r) {
        int m = m0 + r;
        outg[((size_t)b * 64 + m) * 1024 + h * 32 + n] = f2bf(c[r]);
      }
    }
  }
}

// ---------------------------------------------------------------------------
// Host-side orchestration
// ---------------------------------------------------------------------------
extern "C" void kernel_launch(void* const* d_in, const int* in_sizes, int n_in,
                              void* d_out, int out_size, void* d_ws, size_t ws_size,
                              hipStream_t stream) {
  (void)in_sizes; (void)n_in; (void)out_size; (void)ws_size;
  const float* x    = (const float*)d_in[0];
  const float* Wq   = (const float*)d_in[1];
  const float* Wk   = (const float*)d_in[2];
  const float* Wv   = (const float*)d_in[3];
  const float* Wo   = (const float*)d_in[4];
  const float* bo   = (const float*)d_in[5];
  const float* posb = (const float*)d_in[6];
  const float* tcW  = (const float*)d_in[7];
  const float* pcW  = (const float*)d_in[8];
  const float* pcB  = (const float*)d_in[9];
  const float* lnG  = (const float*)d_in[10];
  const float* lnB  = (const float*)d_in[11];
  const float* hpW  = (const float*)d_in[12];
  const float* hpB  = (const float*)d_in[13];
  const float* alpW = (const float*)d_in[14];
  const float* alpB = (const float*)d_in[15];
  float* out = (float*)d_out;

  char* ws = (char*)d_ws;
  size_t off = 0;
  auto take = [&](size_t bytes) -> void* {
    size_t o = off;
    off += (bytes + 255) & ~(size_t)255;
    return (void*)(ws + o);
  };

  u16*   xb      = (u16*)take(67108864ull * 2);   // x bf16 [65536,1024]
  u16*   wqb     = (u16*)take(1048576ull * 2);
  u16*   wkb     = (u16*)take(1048576ull * 2);
  u16*   wvb     = (u16*)take(1048576ull * 2);
  u16*   wob     = (u16*)take(1048576ull * 2);
  u16*   tcb     = (u16*)take(32768ull * 2);
  u16*   pcb     = (u16*)take(524288ull * 2);
  u16*   hpb     = (u16*)take(2097152ull * 2);
  u16*   alpb    = (u16*)take(1048576ull * 2);
  u16*   qb      = (u16*)take(67108864ull * 2);   // [B,S,H*D]
  u16*   kb      = (u16*)take(67108864ull * 2);
  u16*   vb      = (u16*)take(67108864ull * 2);
  u16*   flatb   = (u16*)take(2097152ull * 2);    // [65536,32] == [1024,2048]
  float* pvpre   = (float*)take(262144ull * 4);   // [1024,256]
  u16*   pvb     = (u16*)take(262144ull * 2);     // [1024,256]
  u16*   hvb     = (u16*)take(8388608ull * 2);    // [H,B,256]
  u16*   logitsb = (u16*)take(134217728ull * 2);  // [H,B,64,64]
  u16*   attnb   = xb;  // overlay: xb is dead after Q/K/V/flat GEMMs

  auto conv = [&](const float* src, u16* dst, long long n) {
    int blocks = (int)((n + 1023) / 1024);
    f32_to_bf16_kernel<<<dim3(blocks), dim3(256), 0, stream>>>(src, dst, n);
  };
  conv(x,    xb,   67108864ll);
  conv(Wq,   wqb,  1048576ll);
  conv(Wk,   wkb,  1048576ll);
  conv(Wv,   wvb,  1048576ll);
  conv(Wo,   wob,  1048576ll);
  conv(tcW,  tcb,  32768ll);
  conv(pcW,  pcb,  524288ll);
  conv(hpW,  hpb,  2097152ll);
  conv(alpW, alpb, 1048576ll);

  auto ggrid = [](int M, int N, int batch) {
    return dim3((unsigned)((N + 127) / 128), (unsigned)(M / 128), (unsigned)batch);
  };

  // Q/K/V projections: [65536,1024] = x @ W
  gemm_bf16_kernel<u16, false><<<ggrid(65536, 1024, 1), 256, 0, stream>>>(
      xb, wqb, nullptr, qb, 65536, 1024, 1024, 0, 0, 0, 0);
  gemm_bf16_kernel<u16, false><<<ggrid(65536, 1024, 1), 256, 0, stream>>>(
      xb, wkb, nullptr, kb, 65536, 1024, 1024, 0, 0, 0, 0);
  gemm_bf16_kernel<u16, false><<<ggrid(65536, 1024, 1), 256, 0, stream>>>(
      xb, wvb, nullptr, vb, 65536, 1024, 1024, 0, 0, 0, 0);

  // smolgen token compress: flat = x @ tc_W  -> [65536,32]
  gemm_bf16_kernel<u16, false><<<ggrid(65536, 32, 1), 256, 0, stream>>>(
      xb, tcb, nullptr, flatb, 65536, 32, 1024, 0, 0, 0, 0);

  // position compress + ReLU: pv_pre = relu(flat @ pc_W + pc_b) -> [1024,256] f32
  gemm_bf16_kernel<float, true><<<ggrid(1024, 256, 1), 256, 0, stream>>>(
      flatb, pcb, pcB, pvpre, 1024, 256, 2048, 0, 0, 0, 0);

  // LayerNorm -> pv bf16
  layernorm_kernel<<<dim3(1024), dim3(256), 0, stream>>>(pvpre, lnG, lnB, pvb);

  // per-head projection (batched over H): hv[h] = pv @ hp_W[h] + hp_b[h]
  gemm_bf16_kernel<u16, false><<<ggrid(1024, 256, 32), 256, 0, stream>>>(
      pvb, hpb, hpB, hvb, 1024, 256, 256,
      /*aBatch*/ 0, /*bBatch*/ 65536, /*biasBatch*/ 256, /*cBatch*/ 262144);

  // logits = hv[H*B,256] @ alp_W + alp_b -> bf16 [H,B,4096]
  gemm_bf16_kernel<u16, false><<<ggrid(32768, 4096, 1), 256, 0, stream>>>(
      hvb, alpb, alpB, logitsb, 32768, 4096, 256, 0, 0, 0, 0);

  // fused attention per (b,h)
  attention_kernel<<<dim3(32768), dim3(32), 0, stream>>>(
      qb, kb, vb, posb, logitsb, attnb);

  // output projection: out = attn @ Wo + bo -> f32 [65536,1024]
  gemm_bf16_kernel<float, false><<<ggrid(65536, 1024, 1), 256, 0, stream>>>(
      attnb, wob, bo, out, 65536, 1024, 1024, 0, 0, 0, 0);
}
